// SparseLlamaAttention_687194768160
// MI455X (gfx1250) — matrix operand order repaired and checked by reading, not compiled, not run on hardware
//
#include <hip/hip_runtime.h>

// ---------------------------------------------------------------------------
// SparseLlamaAttention on MI455X (gfx1250):
//   - all GEMM-shaped work on v_wmma_f32_16x16x32_f16 (f32 accumulate)
//   - LDS staging via global_load_async_to_lds_b128 (ASYNCcnt), double-buffered
//   - flash-attention K chunks via TDM tensor_load_to_lds (TENSORcnt)
// ---------------------------------------------------------------------------

typedef _Float16 h16;
typedef __attribute__((ext_vector_type(16))) _Float16 v16h;
typedef __attribute__((ext_vector_type(8)))  _Float16 v8h;
typedef __attribute__((ext_vector_type(8)))  float    v8f;
typedef __attribute__((ext_vector_type(4)))  unsigned v4u;
typedef __attribute__((ext_vector_type(8)))  int      v8i;
typedef __attribute__((ext_vector_type(4)))  int      v4i;

#define NH    32
#define NKVH  8
#define HD    128
#define SEQ   2048
#define BATCH 2
#define TOK   (BATCH * SEQ)
#define HID   4096
#define KVDIM (NKVH * HD)   // 1024

__device__ __forceinline__ v16h mk16(v8h a, v8h b) {
  return __builtin_shufflevector(a, b, 0,1,2,3,4,5,6,7,8,9,10,11,12,13,14,15);
}

__device__ __forceinline__ v8f wmma_f16(v16h a, v16h b, v8f c) {
  return __builtin_amdgcn_wmma_f32_16x16x32_f16(false, a, false, b, (short)0, c,
                                                false, false);
}

// Async global->LDS copy of 16 bytes per lane (tracked by ASYNCcnt).
// LDS byte address = low 32 bits of the generic pointer (ISA aperture rule).
__device__ __forceinline__ void async_ld_b128(void* lds_dst, const void* gsrc) {
  unsigned loff = (unsigned)(unsigned long long)lds_dst;
  asm volatile("global_load_async_to_lds_b128 %0, %1, off"
               :: "v"(loff), "v"(gsrc) : "memory");
}
__device__ __forceinline__ void wait_async_le4() {
  asm volatile("s_wait_asynccnt 4" ::: "memory");
}
__device__ __forceinline__ void wait_async0() {
  asm volatile("s_wait_asynccnt 0" ::: "memory");
}

// ---------------------------------------------------------------------------
// fp32 -> fp16 conversion (grid-stride)
// ---------------------------------------------------------------------------
__global__ void cvt_f32_to_f16(const float* __restrict__ src,
                               h16* __restrict__ dst, size_t n) {
  size_t i = (size_t)blockIdx.x * blockDim.x + threadIdx.x;
  size_t stride = (size_t)gridDim.x * blockDim.x;
  for (; i < n; i += stride) dst[i] = (h16)src[i];
}

// ---------------------------------------------------------------------------
// C[M][N] = A[M][K] * W[N][K]^T   (f16 in, f32 accumulate)
// 128 threads (4 waves); block tile 64x64; wave tile 32x32 (2x2 WMMA).
// Double-buffered LDS, filled with async-to-LDS copies.
// ---------------------------------------------------------------------------
__global__ __launch_bounds__(128)
void gemm_f16(const h16* __restrict__ A, const h16* __restrict__ W,
              h16* __restrict__ Ch, float* __restrict__ Cf,
              int M, int N, int K) {
  __shared__ __align__(16) h16 a_lds[2][64][40];
  __shared__ __align__(16) h16 b_lds[2][64][40];

  const int tid  = threadIdx.x;
  const int lane = tid & 31;
  const int wave = tid >> 5;
  const int hi   = lane >> 4;
  const int ln   = lane & 15;
  const int m0   = blockIdx.y * 64;
  const int n0   = blockIdx.x * 64;
  const int wm   = (wave >> 1) * 32;
  const int wn   = (wave & 1)  * 32;

  const int srow = tid >> 1;          // 0..63
  const int scol = (tid & 1) * 16;    // 0 or 16

  v8f acc[2][2] = {};

  auto stage = [&](int buf, int kk) {
    const h16* ga = A + (size_t)(m0 + srow) * K + kk + scol;
    async_ld_b128(&a_lds[buf][srow][scol],     ga);
    async_ld_b128(&a_lds[buf][srow][scol + 8], ga + 8);
    const h16* gw = W + (size_t)(n0 + srow) * K + kk + scol;
    async_ld_b128(&b_lds[buf][srow][scol],     gw);
    async_ld_b128(&b_lds[buf][srow][scol + 8], gw + 8);
  };

  const int nk = K >> 5;
  stage(0, 0);
  for (int t = 0; t < nk; ++t) {
    const int buf = t & 1;
    if (t + 1 < nk) {
      stage(buf ^ 1, (t + 1) * 32);   // prefetch next slab into other buffer
      wait_async_le4();               // 4 newest in flight; current slab done
    } else {
      wait_async0();
    }
    __syncthreads();

    v16h af[2], bf[2];
#pragma unroll
    for (int i = 0; i < 2; ++i) {
      // A layout: lane=row M; elems 0..7 -> K=hi*8.., 8..15 -> K=16+hi*8..
      const h16* ap = &a_lds[buf][wm + i * 16 + ln][0];
      af[i] = mk16(*(const v8h*)(ap + hi * 8), *(const v8h*)(ap + 16 + hi * 8));
      // B layout: lane=col N; elems 0..7 -> K=hi*8.., 8..15 -> K=16+hi*8..
      const h16* bp = &b_lds[buf][wn + i * 16 + ln][0];
      bf[i] = mk16(*(const v8h*)(bp + hi * 8), *(const v8h*)(bp + 16 + hi * 8));
    }
#pragma unroll
    for (int i = 0; i < 2; ++i)
#pragma unroll
      for (int j = 0; j < 2; ++j)
        acc[i][j] = wmma_f16(af[i], bf[j], acc[i][j]);
    __syncthreads();
  }

  (void)M;
#pragma unroll
  for (int i = 0; i < 2; ++i)
#pragma unroll
    for (int j = 0; j < 2; ++j)
#pragma unroll
      for (int r = 0; r < 8; ++r) {
        int row = m0 + wm + i * 16 + r + hi * 8;
        int col = n0 + wn + j * 16 + ln;
        float v = acc[i][j][r];
        if (Ch) Ch[(size_t)row * N + col] = (h16)v;
        else    Cf[(size_t)row * N + col] = v;
      }
}

// ---------------------------------------------------------------------------
// RoPE in-place on [TOK][nheads*128] f16. positions = token % SEQ.
// ---------------------------------------------------------------------------
__global__ void rope_kernel(h16* __restrict__ x, int nheads) {
  int idx = blockIdx.x * blockDim.x + threadIdx.x;
  int total = TOK * nheads * 64;
  if (idx >= total) return;
  int d  = idx & 63;
  int hh = (idx >> 6) % nheads;
  int t  = idx / (64 * nheads);
  int pos = t % SEQ;
  float inv = __expf(-(float)(2 * d) * (1.0f / 128.0f) * 9.210340371976184f);
  float ang = (float)pos * inv;
  float c, s;
  __sincosf(ang, &s, &c);
  size_t base = (size_t)t * ((size_t)nheads * HD) + (size_t)hh * HD;
  float x0 = (float)x[base + d];
  float x1 = (float)x[base + d + 64];
  x[base + d]      = (h16)(x0 * c - x1 * s);
  x[base + d + 64] = (h16)(x1 * c + x0 * s);
}

// ---------------------------------------------------------------------------
// V[token][kvh*128+d] -> Vt[b][kvh][d][s]
// ---------------------------------------------------------------------------
__global__ void transpose_v(const h16* __restrict__ V, h16* __restrict__ Vt) {
  int idx = blockIdx.x * blockDim.x + threadIdx.x;
  if (idx >= TOK * KVDIM) return;
  int d   = idx & (HD - 1);
  int kvh = (idx >> 7) & (NKVH - 1);
  int t   = idx >> 10;
  int b = t / SEQ, s = t % SEQ;
  Vt[(((size_t)b * NKVH + kvh) * HD + d) * SEQ + s] =
      V[(size_t)t * KVDIM + kvh * HD + d];
}

// ---------------------------------------------------------------------------
// Flash attention: block = (b,h, 64-query tile), 4 waves x 16 query rows.
// K chunk staged by TDM tensor_load_to_lds; Vt chunk by async-to-LDS copies.
// ---------------------------------------------------------------------------
__global__ __launch_bounds__(128)
void flash_attn(const h16* __restrict__ Q, const h16* __restrict__ Kk,
                const h16* __restrict__ Vt, h16* __restrict__ O) {
  __shared__ __align__(16) h16 k_lds[32][136];     // [key][dim], 272B row (padded)
  __shared__ __align__(16) h16 vt_lds[128][40];    // [dim][key]
  __shared__ __align__(16) h16 p_lds[4][16][40];   // per-wave P tile [row][key]

  const int tid  = threadIdx.x;
  const int lane = tid & 31;
  const int wave = tid >> 5;
  const int hi   = lane >> 4;
  const int ln   = lane & 15;

  const int bh  = blockIdx.x;
  const int b   = bh / NH;
  const int h   = bh % NH;
  const int kvh = h / (NH / NKVH);
  const int qb  = blockIdx.y * 64;
  const int qrow0 = qb + wave * 16;

  // Q fragments (A layout), resident: row M = ln
  v16h qf[4];
  {
    const h16* qp = Q + (size_t)(b * SEQ + qrow0 + ln) * (NH * HD) + h * HD;
#pragma unroll
    for (int kt = 0; kt < 4; ++kt)
      qf[kt] = mk16(*(const v8h*)(qp + kt * 32 + hi * 8),
                    *(const v8h*)(qp + kt * 32 + 16 + hi * 8));
  }

  v8f o[8] = {};
  float mrow[8], lrow[8];
#pragma unroll
  for (int r = 0; r < 8; ++r) { mrow[r] = -3.0e30f; lrow[r] = 0.0f; }
  const float scale = 0.08838834764831845f;  // 1/sqrt(128)

  const int kend = qb + 64;
  for (int kc = 0; kc < kend; kc += 32) {
    // ---- stage K chunk (32 keys x 128 dims) ----
#if __has_builtin(__builtin_amdgcn_tensor_load_to_lds) && __has_builtin(__builtin_amdgcn_s_wait_tensorcnt)
    if (wave == 0) {
      // TDM D#: 2D tile 128x32 of 2-byte elems, row stride 1024 elems,
      // LDS padding 4 DWORDs after every 64 DWORDs -> 272B LDS row pitch.
      unsigned long long ga = (unsigned long long)(const void*)
          (Kk + (size_t)(b * SEQ + kc) * KVDIM + kvh * HD);
      unsigned lv = (unsigned)(unsigned long long)(void*)&k_lds[0][0];
      v4u g0;
      g0.x = 1u;                                             // count=1 (valid)
      g0.y = lv;                                             // lds_addr
      g0.z = (unsigned)ga;                                   // global_addr lo
      g0.w = (unsigned)((ga >> 32) & 0x1FFFFFFu) | (2u << 30); // addr hi | type=2
      v8i g1;
      g1[0] = (int)((1u << 16) | (1u << 20) | (5u << 22) | (3u << 25));
              // data_size=2B | pad_enable | pad_interval=64dw | pad_amount=4dw
      g1[1] = (int)(128u << 16);  // tensor_dim0 = 128
      g1[2] = (int)(32u << 16);   // tensor_dim1 = 32
      g1[3] = (int)(128u << 16);  // tile_dim0 = 128
      g1[4] = 32;                 // tile_dim1 = 32, tile_dim2 = 0
      g1[5] = 1024;               // tensor_dim0_stride = 1024 elems
      g1[6] = 0;
      g1[7] = 0;
      v4i z4 = {0, 0, 0, 0};
      v8i z8 = {0, 0, 0, 0, 0, 0, 0, 0};
      __builtin_amdgcn_tensor_load_to_lds(g0, g1, z4, z4, z8, 0);
    }
#else
    {
      int row = tid >> 2;
      int col = (tid & 3) * 32;
      const h16* gk = Kk + (size_t)(b * SEQ + kc + row) * KVDIM + kvh * HD + col;
#pragma unroll
      for (int u = 0; u < 4; ++u)
        async_ld_b128(&k_lds[row][col + u * 8], gk + u * 8);
    }
#endif
    // ---- stage Vt chunk (128 dims x 32 keys) via async-to-LDS ----
    {
      const h16* gv = Vt + (((size_t)b * NKVH + kvh) * HD + tid) * SEQ + kc;
#pragma unroll
      for (int u = 0; u < 4; ++u)
        async_ld_b128(&vt_lds[tid][u * 8], gv + u * 8);
    }
    wait_async0();
#if __has_builtin(__builtin_amdgcn_tensor_load_to_lds) && __has_builtin(__builtin_amdgcn_s_wait_tensorcnt)
    if (wave == 0) __builtin_amdgcn_s_wait_tensorcnt(0);
#endif
    __syncthreads();

    if (kc <= qrow0 + 15) {  // wave-uniform skip of fully-masked chunks
      // ---- S = Q K^T (two 16-key tiles) ----
      v8f s0 = {}, s1 = {};
#pragma unroll
      for (int kt = 0; kt < 4; ++kt) {
        const h16* k0 = &k_lds[ln][kt * 32 + hi * 8];
        const h16* k1 = &k_lds[16 + ln][kt * 32 + hi * 8];
        v16h b0 = mk16(*(const v8h*)k0, *(const v8h*)(k0 + 16));
        v16h b1 = mk16(*(const v8h*)k1, *(const v8h*)(k1 + 16));
        s0 = wmma_f16(qf[kt], b0, s0);
        s1 = wmma_f16(qf[kt], b1, s1);
      }

      // ---- online softmax ----
      float alpha[8];
#pragma unroll
      for (int r = 0; r < 8; ++r) {
        int q = qrow0 + r + hi * 8;
        float v0 = s0[r] * scale, v1 = s1[r] * scale;
        if (kc + ln > q)      v0 = -3.0e30f;
        if (kc + 16 + ln > q) v1 = -3.0e30f;
        float mx = fmaxf(v0, v1);
#pragma unroll
        for (int off = 8; off >= 1; off >>= 1)
          mx = fmaxf(mx, __shfl_xor(mx, off, 32));
        float mn = fmaxf(mrow[r], mx);
        alpha[r] = __expf(mrow[r] - mn);
        mrow[r] = mn;
        v0 = __expf(v0 - mn);
        v1 = __expf(v1 - mn);
        float sum = v0 + v1;
#pragma unroll
        for (int off = 8; off >= 1; off >>= 1)
          sum += __shfl_xor(sum, off, 32);
        lrow[r] = lrow[r] * alpha[r] + sum;
        p_lds[wave][r + hi * 8][ln]      = (h16)v0;
        p_lds[wave][r + hi * 8][16 + ln] = (h16)v1;
      }
#pragma unroll
      for (int t = 0; t < 8; ++t)
#pragma unroll
        for (int r = 0; r < 8; ++r)
          o[t][r] *= alpha[r];

      // same-wave cross-lane LDS RAW: drain DS pipe before reloading P
      asm volatile("s_wait_dscnt 0" ::: "memory");

      // ---- O += P V ----
      const h16* pp = &p_lds[wave][ln][0];
      v16h pf = mk16(*(const v8h*)(pp + hi * 8), *(const v8h*)(pp + 16 + hi * 8));
#pragma unroll
      for (int nt = 0; nt < 8; ++nt) {
        const h16* vp = &vt_lds[nt * 16 + ln][hi * 8];
        v16h vf = mk16(*(const v8h*)vp, *(const v8h*)(vp + 16));
        o[nt] = wmma_f16(pf, vf, o[nt]);
      }
    }
    __syncthreads();
  }

  // ---- normalize, store attn output f16 [token][h*128+d] ----
#pragma unroll
  for (int r = 0; r < 8; ++r) {
    float inv = 1.0f / lrow[r];
    int row = qrow0 + r + hi * 8;
#pragma unroll
    for (int nt = 0; nt < 8; ++nt)
      O[(size_t)(b * SEQ + row) * (NH * HD) + h * HD + nt * 16 + ln] =
          (h16)(o[nt][r] * inv);
  }
}

// ---------------------------------------------------------------------------
// host-side launcher
// ---------------------------------------------------------------------------
extern "C" void kernel_launch(void* const* d_in, const int* in_sizes, int n_in,
                              void* d_out, int out_size, void* d_ws,
                              size_t ws_size, hipStream_t stream) {
  (void)in_sizes; (void)n_in; (void)out_size; (void)ws_size;
  const float* hs = (const float*)d_in[0];
  const float* Wq = (const float*)d_in[1];
  const float* Wk = (const float*)d_in[2];
  const float* Wv = (const float*)d_in[3];
  const float* Wo = (const float*)d_in[4];
  float* out = (float*)d_out;

  char* ws = (char*)d_ws;
  size_t off = 0;
  auto alloc16 = [&](size_t elems) {
    h16* p = (h16*)(ws + off);
    off += ((elems * sizeof(h16) + 255) / 256) * 256;
    return p;
  };
  h16* hs16 = alloc16((size_t)TOK * HID);
  h16* wq16 = alloc16((size_t)HID * HID);
  h16* wk16 = alloc16((size_t)KVDIM * HID);
  h16* wv16 = alloc16((size_t)KVDIM * HID);
  h16* wo16 = alloc16((size_t)HID * HID);
  h16* q16  = alloc16((size_t)TOK * HID);
  h16* k16  = alloc16((size_t)TOK * KVDIM);
  h16* v16  = alloc16((size_t)TOK * KVDIM);
  h16* vt16 = alloc16((size_t)TOK * KVDIM);
  h16* at16 = alloc16((size_t)TOK * HID);

  cvt_f32_to_f16<<<2048, 256, 0, stream>>>(hs, hs16, (size_t)TOK * HID);
  cvt_f32_to_f16<<<2048, 256, 0, stream>>>(Wq, wq16, (size_t)HID * HID);
  cvt_f32_to_f16<<<1024, 256, 0, stream>>>(Wk, wk16, (size_t)KVDIM * HID);
  cvt_f32_to_f16<<<1024, 256, 0, stream>>>(Wv, wv16, (size_t)KVDIM * HID);
  cvt_f32_to_f16<<<2048, 256, 0, stream>>>(Wo, wo16, (size_t)HID * HID);

  gemm_f16<<<dim3(HID / 64,   TOK / 64), 128, 0, stream>>>(hs16, wq16, q16, nullptr, TOK, HID,   HID);
  gemm_f16<<<dim3(KVDIM / 64, TOK / 64), 128, 0, stream>>>(hs16, wk16, k16, nullptr, TOK, KVDIM, HID);
  gemm_f16<<<dim3(KVDIM / 64, TOK / 64), 128, 0, stream>>>(hs16, wv16, v16, nullptr, TOK, KVDIM, HID);

  rope_kernel<<<(TOK * NH * 64 + 255) / 256, 256, 0, stream>>>(q16, NH);
  rope_kernel<<<(TOK * NKVH * 64 + 255) / 256, 256, 0, stream>>>(k16, NKVH);

  transpose_v<<<(TOK * KVDIM + 255) / 256, 256, 0, stream>>>(v16, vt16);

  flash_attn<<<dim3(BATCH * NH, SEQ / 64), 128, 0, stream>>>(q16, k16, vt16, at16);

  gemm_f16<<<dim3(HID / 64, TOK / 64), 128, 0, stream>>>(at16, wo16, nullptr, out, TOK, HID, HID);
}
